// SparseAutoencoder_11089605558510
// MI455X (gfx1250) — matrix operand order repaired and checked by reading, not compiled, not run on hardware
//
#include <hip/hip_runtime.h>

// ---------------------------------------------------------------------------
// SparseAutoencoder top-k for MI455X (gfx1250, wave32, WMMA + TDM)
//   pre  = (x - b_dec) @ W_enc + b_enc     [2048 x 32768]
//   acts = relu(pre); keep top-64 per row, zero the rest.
// GEMM: v_wmma_f32_16x16x32_bf16, tiles staged by tensor_load_to_lds (TDM),
// double-buffered with s_wait_tensorcnt.
// ---------------------------------------------------------------------------

#define N_ROWS   2048
#define D_MODEL  4096
#define D_HIDDEN 32768

typedef __attribute__((ext_vector_type(16))) __bf16 v16bf;
typedef __attribute__((ext_vector_type(8)))  float  v8f;
typedef __attribute__((ext_vector_type(4)))  unsigned int v4u;
typedef __attribute__((ext_vector_type(8)))  int    v8i;
typedef __attribute__((ext_vector_type(4)))  int    v4i;

union FragCast { uint4 u[2]; v16bf v; };   // 32 bytes

static __device__ __forceinline__ unsigned short f32_to_bf16(float f) {
    unsigned u = __float_as_uint(f);
    unsigned r = u + 0x7FFFu + ((u >> 16) & 1u);   // round-to-nearest-even
    return (unsigned short)(r >> 16);
}

// ---------------------------------------------------------------------------
// TDM: async-load a 2D bf16 tile (tile_d0 contiguous elems x tile_d1 rows)
// from global into LDS, padding each 64B row with 16B (-> 40-elem LDS rows).
// D# layout per CDNA5 ISA 8.3/8.4. Tracked by TENSORcnt.
// ---------------------------------------------------------------------------
static __device__ __forceinline__ void tdm_load_tile_bf16(
        unsigned lds_addr, const unsigned short* gptr,
        unsigned tensor_d0, unsigned tensor_d1, unsigned d0_stride,
        unsigned tile_d0, unsigned tile_d1) {
    unsigned long long ga = (unsigned long long)(uintptr_t)gptr;
    v4u g0;
    g0[0] = 1u;                                         // count=1 (valid D#)
    g0[1] = lds_addr;                                   // [63:32] LDS byte addr
    g0[2] = (unsigned)(ga & 0xFFFFFFFFu);               // [95:64] global addr lo
    g0[3] = (unsigned)((ga >> 32) & 0x01FFFFFFu)        // [120:96] global addr hi
          | (2u << 30);                                 // [127:126] type=2 (image)
    v8i g1;
    g1[0] = (int)((1u << 16)        // data_size = 1 -> 2-byte elements
                | (1u << 20)        // pad_enable
                | (3u << 22)        // pad_interval = 3 -> every 16 DWORDs (64 B row)
                | (3u << 25));      // pad_amount   = 3 -> 4 DWORDs (16 B)
    g1[1] = (int)((tensor_d0 & 0xFFFFu) << 16);         // [63:48] tensor_dim0 lo16
    g1[2] = (int)(((tensor_d0 >> 16) & 0xFFFFu)         // [79:64] tensor_dim0 hi16
                | ((tensor_d1 & 0xFFFFu) << 16));       // [95:80] tensor_dim1 lo16
    g1[3] = (int)(((tensor_d1 >> 16) & 0xFFFFu)         // [111:96] tensor_dim1 hi16
                | ((tile_d0 & 0xFFFFu) << 16));         // [127:112] tile_dim0
    g1[4] = (int)(tile_d1 & 0xFFFFu);                   // [143:128] tile_dim1, tile_dim2=0
    g1[5] = (int)d0_stride;                             // [191:160] tensor_dim0_stride lo32
    g1[6] = 0;                                          // stride hi / dim1_stride
    g1[7] = 0;
    v4i g2 = {0, 0, 0, 0};                              // 2D tile: dims 2+ unused
    v4i g3 = {0, 0, 0, 0};
#if __clang_major__ >= 23
    v8i g4 = {0, 0, 0, 0, 0, 0, 0, 0};
    __builtin_amdgcn_tensor_load_to_lds(g0, g1, g2, g3, g4, 0);
#else
    __builtin_amdgcn_tensor_load_to_lds(g0, g1, g2, g3, 0);
#endif
}

// ---------------------------------------------------------------------------
// Kernel 1: xc[m][k] = bf16(x[m][k] - b_dec[k])        (16 MB into ws)
// ---------------------------------------------------------------------------
__global__ __launch_bounds__(256) void convert_x(const float* __restrict__ x,
                                                 const float* __restrict__ b_dec,
                                                 unsigned short* __restrict__ xc) {
    size_t i = (((size_t)blockIdx.x << 8) + threadIdx.x) << 2;   // 4 elems/thread
    int kcol = (int)(i & (D_MODEL - 1));
    float4 a = *(const float4*)(x + i);
    float4 b = *(const float4*)(b_dec + kcol);
    ushort4 o;
    o.x = f32_to_bf16(a.x - b.x);
    o.y = f32_to_bf16(a.y - b.y);
    o.z = f32_to_bf16(a.z - b.z);
    o.w = f32_to_bf16(a.w - b.w);
    *(ushort4*)(xc + i) = o;
}

// ---------------------------------------------------------------------------
// Kernel 2: wt[n][k] = bf16(W_enc[k][n])   transpose via LDS (256 MB into ws)
// ---------------------------------------------------------------------------
__global__ __launch_bounds__(256) void convert_wt(const float* __restrict__ W,
                                                  unsigned short* __restrict__ wt) {
    __shared__ float tile[32][33];
    int tx = threadIdx.x & 31, ty = threadIdx.x >> 5;       // 32 x 8
    int n0 = blockIdx.x * 32, k0 = blockIdx.y * 32;
#pragma unroll
    for (int j = 0; j < 4; ++j)
        tile[ty + j * 8][tx] = W[(size_t)(k0 + ty + j * 8) * D_HIDDEN + n0 + tx];
    __syncthreads();
#pragma unroll
    for (int j = 0; j < 4; ++j)
        wt[(size_t)(n0 + ty + j * 8) * D_MODEL + k0 + tx] =
            f32_to_bf16(tile[tx][ty + j * 8]);
}

// ---------------------------------------------------------------------------
// Kernel 3: acts = relu(xc @ wt^T + b_enc) -> d_out (dense fp32)
//   WG tile 128(M) x 256(N), 8 waves (2x4) of 64x64, BK = 32.
//   Double-buffered TDM staging; fragment layouts per CDNA5 ISA 7.12.2.
// ---------------------------------------------------------------------------
#define BM 128
#define BN 256
#define BK 32
#define LDP (BK + 8)    // 40-elem LDS rows (80 B): bank-spread, 16B aligned;
                        // produced directly by the TDM pad feature.

__global__ __launch_bounds__(256) void gemm_bias_relu(
        const unsigned short* __restrict__ xc,   // [N_ROWS][D_MODEL] bf16
        const unsigned short* __restrict__ wt,   // [D_HIDDEN][D_MODEL] bf16
        const float* __restrict__ b_enc,
        float* __restrict__ out) {
    __shared__ unsigned short As[2][BM * LDP];
    __shared__ unsigned short Bs[2][BN * LDP];

    const int tid  = threadIdx.x;
    const int lane = tid & 31;
    const int wave = tid >> 5;
    const int wm   = wave & 1;          // 2 waves along M
    const int wn   = wave >> 1;         // 4 waves along N
    const int half = lane >> 4;         // lane half (ISA layout)
    const int lr   = lane & 15;
    const int m0   = blockIdx.y * BM;
    const int n0   = blockIdx.x * BN;
    const bool issuer = (wave == 0);

    const unsigned short* aG = xc + (size_t)m0 * D_MODEL;
    const unsigned short* bG = wt + (size_t)n0 * D_MODEL;
    const unsigned ldsA0 = (unsigned)(uintptr_t)&As[0][0];
    const unsigned ldsA1 = (unsigned)(uintptr_t)&As[1][0];
    const unsigned ldsB0 = (unsigned)(uintptr_t)&Bs[0][0];
    const unsigned ldsB1 = (unsigned)(uintptr_t)&Bs[1][0];

    v8f acc[4][4] = {};

    // ---- prologue: DMA first K-tiles into buffer 0
    if (issuer) {
        tdm_load_tile_bf16(ldsA0, aG, D_MODEL, N_ROWS,   D_MODEL, BK, BM);
        tdm_load_tile_bf16(ldsB0, bG, D_MODEL, D_HIDDEN, D_MODEL, BK, BN);
    }

    int buf = 0;
    for (int kk = 0; kk < D_MODEL; kk += BK, buf ^= 1) {
        if (issuer) {
            int kn = kk + BK;
            if (kn < D_MODEL) {
                // prefetch next K-tiles into the other buffer
                tdm_load_tile_bf16(buf ? ldsA0 : ldsA1, aG + kn,
                                   D_MODEL, N_ROWS,   D_MODEL, BK, BM);
                tdm_load_tile_bf16(buf ? ldsB0 : ldsB1, bG + kn,
                                   D_MODEL, D_HIDDEN, D_MODEL, BK, BN);
                __builtin_amdgcn_s_wait_tensorcnt(2);   // current pair done
            } else {
                __builtin_amdgcn_s_wait_tensorcnt(0);
            }
        }
        __syncthreads();   // publish current buffer to all waves

        const unsigned short* Ab = &As[buf][0];
        const unsigned short* Bb = &Bs[buf][0];

        // ---- A fragments: lane holds M = lr, K runs {half*8..+7}, {16+half*8..+7}
        v16bf af[4];
#pragma unroll
        for (int mi = 0; mi < 4; ++mi) {
            int m = wm * 64 + mi * 16 + lr;
            FragCast fc;
            fc.u[0] = *(const uint4*)(Ab + m * LDP + half * 8);
            fc.u[1] = *(const uint4*)(Ab + m * LDP + 16 + half * 8);
            af[mi] = fc.v;
        }
        // ---- B fragments: lane holds N = lr, K runs {half*8..+7}, {16+half*8..+7}
#pragma unroll
        for (int ni = 0; ni < 4; ++ni) {
            int n = wn * 64 + ni * 16 + lr;
            FragCast fc;
            fc.u[0] = *(const uint4*)(Bb + n * LDP + half * 8);
            fc.u[1] = *(const uint4*)(Bb + n * LDP + 16 + half * 8);
            v16bf bfr = fc.v;
#pragma unroll
            for (int mi = 0; mi < 4; ++mi) {
                acc[mi][ni] = __builtin_amdgcn_wmma_f32_16x16x32_bf16(
                    false, af[mi], false, bfr, (short)0, acc[mi][ni], false, false);
            }
        }
        __syncthreads();   // release buffer before next iteration's DMA reuses it
    }

    // ---- epilogue: +bias, relu, store.  C layout: lane half->M offset 8, c[r]->M+r
#pragma unroll
    for (int ni = 0; ni < 4; ++ni) {
        int n = n0 + wn * 64 + ni * 16 + lr;
        float bias = b_enc[n];
#pragma unroll
        for (int mi = 0; mi < 4; ++mi) {
            int mbase = m0 + wm * 64 + mi * 16 + half * 8;
#pragma unroll
            for (int r = 0; r < 8; ++r) {
                float v = acc[mi][ni][r] + bias;
                out[(size_t)(mbase + r) * D_HIDDEN + n] = v > 0.f ? v : 0.f;
            }
        }
    }
}

// ---------------------------------------------------------------------------
// Kernel 4: per-row exact top-k via 3-level radix select on float bits
//   (acts >= 0 after relu, so uint compare of raw bits is monotonic).
//   Level A: bits 31..19 | Level B: bits 18..6 | Level C: bits 5..0.
// ---------------------------------------------------------------------------
__global__ __launch_bounds__(256) void topk_select(float* __restrict__ acts,
                                                   const int* __restrict__ kptr) {
    __shared__ unsigned hist[8192];          // 32 KB
    __shared__ unsigned sBin, sAbove, sT;
    __shared__ int sEq;

    const int tid = threadIdx.x;
    const int kv  = *kptr;
    float* p = acts + (size_t)blockIdx.x * D_HIDDEN;

    // ---------- level A ----------
    for (int i = tid; i < 8192; i += 256) hist[i] = 0u;
    __syncthreads();
    for (int i = 0; i < 128; ++i) {
        unsigned key = __float_as_uint(p[tid + (i << 8)]);
        atomicAdd(&hist[key >> 19], 1u);
    }
    __syncthreads();
    if (tid == 0) {
        unsigned cum = 0; int b = 8191;
        while (b >= 0) { cum += hist[b]; if (cum >= (unsigned)kv) break; --b; }
        if (b < 0) b = 0;
        sBin = (unsigned)b; sAbove = cum - hist[b];
    }
    __syncthreads();
    unsigned b0 = sBin, above = sAbove;

    // ---------- level B ----------
    __syncthreads();
    for (int i = tid; i < 8192; i += 256) hist[i] = 0u;
    __syncthreads();
    for (int i = 0; i < 128; ++i) {
        unsigned key = __float_as_uint(p[tid + (i << 8)]);
        if ((key >> 19) == b0) atomicAdd(&hist[(key >> 6) & 0x1FFFu], 1u);
    }
    __syncthreads();
    if (tid == 0) {
        unsigned need = (unsigned)kv - above;
        unsigned cum = 0; int b = 8191;
        while (b >= 0) { cum += hist[b]; if (cum >= need) break; --b; }
        if (b < 0) b = 0;
        sBin = (unsigned)b; sAbove = above + (cum - hist[b]);
    }
    __syncthreads();
    unsigned b1 = sBin; above = sAbove;
    unsigned pre26 = (b0 << 13) | b1;     // top 26 bits of threshold prefix

    // ---------- level C ----------
    __syncthreads();
    for (int i = tid; i < 64; i += 256) hist[i] = 0u;
    __syncthreads();
    for (int i = 0; i < 128; ++i) {
        unsigned key = __float_as_uint(p[tid + (i << 8)]);
        if ((key >> 6) == pre26) atomicAdd(&hist[key & 0x3Fu], 1u);
    }
    __syncthreads();
    if (tid == 0) {
        unsigned need = (unsigned)kv - above;
        unsigned cum = 0; int b = 63;
        while (b >= 0) { cum += hist[b]; if (cum >= need) break; --b; }
        if (b < 0) b = 0;
        unsigned nGt = above + (cum - hist[b]);          // strictly > threshold
        sT  = (pre26 << 6) | (unsigned)b;                // exact threshold key
        sEq = (int)((unsigned)kv - nGt);                 // ties to keep
    }
    __syncthreads();
    unsigned T = sT;

    // ---------- rewrite row in place ----------
    for (int i = 0; i < 128; ++i) {
        int idx = tid + (i << 8);
        float v = p[idx];
        unsigned key = __float_as_uint(v);
        float o = 0.0f;
        if (key > T) {
            o = v;
        } else if (key == T) {
            if (atomicSub(&sEq, 1) > 0) o = v;
        }
        p[idx] = o;
    }
}

// ---------------------------------------------------------------------------
// Launch. d_ws layout: [0, 16MB) xc bf16, [16MB, 272MB) wt bf16.
// ---------------------------------------------------------------------------
extern "C" void kernel_launch(void* const* d_in, const int* in_sizes, int n_in,
                              void* d_out, int out_size, void* d_ws, size_t ws_size,
                              hipStream_t stream) {
    const float* x     = (const float*)d_in[0];
    const float* W_enc = (const float*)d_in[1];
    const float* b_enc = (const float*)d_in[2];
    const float* b_dec = (const float*)d_in[3];
    const int*   kptr  = (const int*)d_in[4];

    float* out = (float*)d_out;
    unsigned short* xc = (unsigned short*)d_ws;
    unsigned short* wt = (unsigned short*)((char*)d_ws + (size_t)N_ROWS * D_MODEL * 2);

    // 1) x - b_dec -> bf16
    convert_x<<<(N_ROWS * D_MODEL) / (256 * 4), 256, 0, stream>>>(x, b_dec, xc);
    // 2) W_enc^T -> bf16 (K-contiguous rows for WMMA B fragments)
    convert_wt<<<dim3(D_HIDDEN / 32, D_MODEL / 32), 256, 0, stream>>>(W_enc, wt);
    // 3) WMMA GEMM (TDM-staged, double-buffered) + bias + relu -> d_out
    gemm_bias_relu<<<dim3(D_HIDDEN / BN, N_ROWS / BM), 256, 0, stream>>>(xc, wt, b_enc, out);
    // 4) per-row exact top-k, zero the rest (in place)
    topk_select<<<N_ROWS, 256, 0, stream>>>(out, kptr);
}
